// EncoderDecoderLSTM_69106023792936
// MI455X (gfx1250) — matrix operand order repaired and checked by reading, not compiled, not run on hardware
//
#include <hip/hip_runtime.h>
#include <hip/hip_bf16.h>

// ---------------------------------------------------------------------------
// EncoderDecoderLSTM for MI455X (gfx1250, wave32, WMMA).
// B=64, T=128, F_IN=12, H=128.  bf16 WMMA (f32 accum) for all K=128 GEMMs.
// Gate-aligned N-tiling: each wave owns all 4 gates of a 16x16 (b,h) patch,
// so LSTM gate math runs on WMMA accumulators in registers (no LDS exchange).
// B-fragments are double-buffered from LDS with per-gate base pointers so the
// WMMA stream has no address VALU (hazard NOPs) and overlapped ds loads.
// ---------------------------------------------------------------------------

#define BSZ   64
#define TSEQ  128
#define HDIM  128
#define GDIM  512      // 4*H
#define FIN   12

typedef __attribute__((ext_vector_type(16))) __bf16 v16bf;
typedef __attribute__((ext_vector_type(8)))  float  v8f;

union ABu { v16bf v; unsigned x[8]; };   // x[j] = packed bf16 pair (K, K+1)
union Du  { v8f   v; float   f[8]; };

__device__ __forceinline__ unsigned short f2bf(float f) {
  unsigned u = __float_as_uint(f);
  unsigned r = (u + 0x7FFFu + ((u >> 16) & 1u)) >> 16;   // RNE
  return (unsigned short)r;
}
__device__ __forceinline__ unsigned pack2bf(float a, float b) {
  return (unsigned)f2bf(a) | ((unsigned)f2bf(b) << 16);
}
__device__ __forceinline__ float sigm(float x) { return 1.0f / (1.0f + __expf(-x)); }

__device__ __forceinline__ v8f wmma_bf(v16bf a, v16bf b, v8f c) {
  // D = A(16x32 bf16) x B(32x16 bf16) + C(16x16 f32)
  return __builtin_amdgcn_wmma_f32_16x16x32_bf16(false, a, false, b, (short)0, c, false, false);
}

// Grid-wide sense barrier (counters zeroed by prep kernel each launch).
__device__ __forceinline__ void gbar(unsigned* cnt, unsigned* gen, unsigned nwg, unsigned& phase) {
  __threadfence();
  __syncthreads();
  if (threadIdx.x == 0) {
    unsigned prev = atomicAdd(cnt, 1u);
    if (prev == phase * nwg + (nwg - 1u)) {
      atomicAdd(gen, 1u);
    } else {
      while (atomicAdd(gen, 0u) <= phase) __builtin_amdgcn_s_sleep(1);
    }
  }
  __syncthreads();
  phase++;
}

// ---------------------------------------------------------------------------
// prep: f32 -> bf16 weight conversion + barrier init
// ---------------------------------------------------------------------------
__global__ void prep_kernel(const float* __restrict__ encWhh, const float* __restrict__ decWhh,
                            const float* __restrict__ attnW,
                            unsigned short* WhhE, unsigned short* WhhD,
                            unsigned short* W1B, unsigned short* W2B, unsigned* bar) {
  int g = blockIdx.x * blockDim.x + threadIdx.x;
  if (g == 0) { bar[0] = 0u; bar[1] = 0u; }
  if (g < 65536) {
    WhhE[g] = f2bf(encWhh[g]);
  } else if (g < 131072) {
    int q = g - 65536;  WhhD[q] = f2bf(decWhh[q]);
  } else if (g < 147456) {
    int q = g - 131072; W1B[q] = f2bf(attnW[(q >> 7) * 2 * HDIM + (q & 127)]);
  } else if (g < 163840) {
    int q = g - 147456; W2B[q] = f2bf(attnW[(q >> 7) * 2 * HDIM + HDIM + (q & 127)]);
  }
}

// ---------------------------------------------------------------------------
// Lane geometry for 16x16 tiles (ISA VGPR layouts).
//   wave -> mtile = wave>>3 (batch rows), n0 = wave&7 (h columns)
//   gate g lives at Whh rows g*128 + n0*16 + ncol.
// ---------------------------------------------------------------------------
struct LaneGeom {
  int mrow;    // A-frag source row in hS
  int kA;      // A lane-half K offset (0/8)
  int ncol;    // 0..15
  int kB;      // B lane-half K offset (0/8)
  int drow0;   // 0/8
  int hh;      // this lane's h column
  int m0;      // first batch row of this lane's D rows
};

__device__ __forceinline__ LaneGeom lane_geom(int wave, int lane) {
  LaneGeom g;
  const int mtile = wave >> 3, n0 = wave & 7;
  g.mrow  = mtile * 16 + (lane & 15);
  g.kA    = (lane & 16) ? 8 : 0;
  g.ncol  = lane & 15;
  g.kB    = (lane & 16) ? 8 : 0;
  g.drow0 = (lane & 16) ? 8 : 0;
  g.hh    = n0 * 16 + g.ncol;
  g.m0    = mtile * 16 + g.drow0;
  return g;
}

// Load the 4 A-fragments (K=0..127) for this wave's mtile from hS.
__device__ __forceinline__ void load_a_frags(const unsigned short* hS, const LaneGeom& gm, ABu a[4]) {
  #pragma unroll
  for (int kc = 0; kc < 4; kc++)
    #pragma unroll
    for (int j = 0; j < 8; j++) {
      int k = kc * 32 + gm.kA + ((j & 4) ? 16 : 0) + 2 * (j & 3);
      a[kc].x[j] = *(const unsigned*)&hS[gm.mrow * HDIM + k];
    }
}

// Fetch one 32x16 B fragment: base points at (row, kB); kc selects K-chunk.
__device__ __forceinline__ void load_b_frag(const unsigned short* base, int kc, ABu& bb) {
  #pragma unroll
  for (int j = 0; j < 8; j++)
    bb.x[j] = *(const unsigned*)(base + kc * 32 + ((j & 4) ? 16 : 0) + 2 * (j & 3));
}

// z-accumulators for the 4 gates of this wave's patch. Per-gate LDS base
// pointers keep every ds offset within the 16-bit immediate; B fragments are
// double-buffered so ds loads overlap the WMMA pipe.
__device__ __forceinline__ void gate_gemm(const unsigned short* WS, const LaneGeom& gm,
                                          int n0, const ABu a[4], v8f acc[4]) {
  const unsigned short* p[4];
  #pragma unroll
  for (int g = 0; g < 4; g++)
    p[g] = WS + (g * HDIM + n0 * 16 + gm.ncol) * HDIM + gm.kB;

  ABu buf[2];
  load_b_frag(p[0], 0, buf[0]);
  #pragma unroll
  for (int it = 0; it < 16; it++) {
    const int g = it >> 2, kc = it & 3;
    const int nit = it + 1;
    if (nit < 16) load_b_frag(p[nit >> 2], nit & 3, buf[nit & 1]);
    if (kc == 0) {
      #pragma unroll
      for (int r = 0; r < 8; r++) acc[g][r] = 0.0f;
    }
    acc[g] = wmma_bf(a[kc].v, buf[it & 1].v, acc[g]);
  }
}

// ---------------------------------------------------------------------------
// Encoder: single persistent workgroup, 128 steps
// LDS: Whh bf16 128KB + h bf16 16KB + x stage 3KB = 147KB
// ---------------------------------------------------------------------------
__global__ __launch_bounds__(1024) void enc_kernel(const float* __restrict__ x,
                                                   const float* __restrict__ Wih,
                                                   const float* __restrict__ bias,
                                                   const unsigned short* __restrict__ WhhB,
                                                   float* __restrict__ enc_out,
                                                   float* __restrict__ cT) {
  __shared__ unsigned short WS[GDIM * HDIM];    // 128 KB, Whh bf16
  __shared__ unsigned short hS[BSZ * HDIM];     // 16 KB
  __shared__ float          xtS[BSZ * FIN];     // 3 KB
  const int tid = threadIdx.x, lane = tid & 31, wave = tid >> 5;
  const LaneGeom gm = lane_geom(wave, lane);
  const int n0 = wave & 7;

  // stage Whh into LDS (dword copies), zero h
  for (int i = tid; i < GDIM * HDIM / 2; i += 1024)
    ((unsigned*)WS)[i] = ((const unsigned*)WhhB)[i];
  for (int i = tid; i < BSZ * HDIM; i += 1024) hS[i] = 0;

  float creg[8];
  #pragma unroll
  for (int r = 0; r < 8; r++) creg[r] = 0.0f;

  // Register-cache the 4 Wih gate rows + biases for this lane's h column.
  float wih_c[4][FIN], b_c[4];
  #pragma unroll
  for (int g = 0; g < 4; g++) {
    b_c[g] = bias[g * HDIM + gm.hh];
    for (int k = 0; k < FIN; k++) wih_c[g][k] = Wih[(g * HDIM + gm.hh) * FIN + k];
  }

  for (int t = 0; t < TSEQ; t++) {
    __syncthreads();                       // prev-iter hS/xtS consumers done
    if (tid < BSZ * FIN) xtS[tid] = x[((tid / FIN) * TSEQ + t) * FIN + (tid % FIN)];
    ABu a[4];
    load_a_frags(hS, gm, a);
    __syncthreads();                       // A reads + xt stage complete
    asm volatile("" ::: "memory");         // keep B-frag LDS loads inside loop (no spill-hoist)

    v8f acc[4];
    gate_gemm(WS, gm, n0, a, acc);

    Du dI, dF, dG, dO;
    dI.v = acc[0]; dF.v = acc[1]; dG.v = acc[2]; dO.v = acc[3];
    #pragma unroll
    for (int r = 0; r < 8; r++) {
      const int b = gm.m0 + r;
      const float* xr = &xtS[b * FIN];
      float s0 = b_c[0], s1 = b_c[1], s2 = b_c[2], s3 = b_c[3];
      #pragma unroll
      for (int k = 0; k < FIN; k++) {
        float xv = xr[k];
        s0 += xv * wih_c[0][k]; s1 += xv * wih_c[1][k];
        s2 += xv * wih_c[2][k]; s3 += xv * wih_c[3][k];
      }
      float zi = dI.f[r] + s0, zf = dF.f[r] + s1;
      float zg = dG.f[r] + s2, zo = dO.f[r] + s3;
      float cn = sigm(zf) * creg[r] + sigm(zi) * tanhf(zg);
      float hn = sigm(zo) * tanhf(cn);
      creg[r] = cn;
      hS[b * HDIM + gm.hh] = f2bf(hn);
      enc_out[(b * TSEQ + t) * HDIM + gm.hh] = hn;
    }
  }
  #pragma unroll
  for (int r = 0; r < 8; r++) cT[(gm.m0 + r) * HDIM + gm.hh] = creg[r];
}

// ---------------------------------------------------------------------------
// enc_proj = enc_out @ W2^T + attn_b    (M=8192, N=128, K=128)
// ---------------------------------------------------------------------------
__global__ __launch_bounds__(256) void proj_kernel(const float* __restrict__ enc_out,
                                                   const unsigned short* __restrict__ W2B,
                                                   const float* __restrict__ attn_b,
                                                   float* __restrict__ enc_proj) {
  const int tid = threadIdx.x, lane = tid & 31, wave = tid >> 5;
  const int mtile = blockIdx.x;            // 512 blocks
  const int ntile = wave;                  // 8 waves
  const int mrow  = mtile * 16 + (lane & 15);
  const int kA    = (lane & 16) ? 8 : 0;
  const int ncol  = lane & 15;
  const int kB    = (lane & 16) ? 8 : 0;
  const int drow0 = (lane & 16) ? 8 : 0;

  float bias = attn_b[ntile * 16 + ncol];
  v8f acc;
  #pragma unroll
  for (int r = 0; r < 8; r++) acc[r] = bias;
  const unsigned short* wbase = W2B + (ntile * 16 + ncol) * HDIM + kB;
  #pragma unroll
  for (int kc = 0; kc < 4; kc++) {
    ABu a, bb;
    #pragma unroll
    for (int j = 0; j < 8; j++) {
      int k = kc * 32 + kA + ((j & 4) ? 16 : 0) + 2 * (j & 3);
      const float* s = &enc_out[mrow * HDIM + k];
      a.x[j] = pack2bf(s[0], s[1]);
    }
    load_b_frag(wbase, kc, bb);
    acc = wmma_bf(a.v, bb.v, acc);
  }
  Du d; d.v = acc;
  #pragma unroll
  for (int r = 0; r < 8; r++)
    enc_proj[(mtile * 16 + drow0 + r) * HDIM + ntile * 16 + ncol] = d.f[r];
}

// ---------------------------------------------------------------------------
// Decoder + attention: persistent 16-workgroup grid with global barriers.
// wg0 carries the LSTM recurrence + q GEMM (WMMA); all wgs do attention.
// ---------------------------------------------------------------------------
__global__ __launch_bounds__(1024) void dec_kernel(const float* __restrict__ x,
                                                   const float* __restrict__ Wih,
                                                   const float* __restrict__ bias,
                                                   const unsigned short* __restrict__ WhhB,
                                                   const unsigned short* __restrict__ W1B,
                                                   const float* __restrict__ enc_out,
                                                   const float* __restrict__ enc_proj,
                                                   const float* __restrict__ cT,
                                                   const float* __restrict__ v_w,
                                                   float* __restrict__ qbuf,
                                                   float* __restrict__ score,
                                                   float* __restrict__ dec_cat,
                                                   unsigned* bar) {
  __shared__ unsigned short WS[GDIM * HDIM];
  __shared__ unsigned short hS[BSZ * HDIM];
  __shared__ float          xtS[BSZ * FIN];
  const int tid = threadIdx.x, lane = tid & 31, wave = tid >> 5, blk = blockIdx.x;
  const LaneGeom gm = lane_geom(wave, lane);
  const int n0 = wave & 7;
  unsigned* cnt = bar;
  unsigned* gen = bar + 1;
  unsigned  phase = 0;

  float creg[8], wih_c[4][FIN], b_c[4];
  if (blk == 0) {
    for (int i = tid; i < GDIM * HDIM / 2; i += 1024)
      ((unsigned*)WS)[i] = ((const unsigned*)WhhB)[i];
    for (int i = tid; i < BSZ * HDIM; i += 1024)
      hS[i] = f2bf(enc_out[((i >> 7) * TSEQ + (TSEQ - 1)) * HDIM + (i & 127)]);
    #pragma unroll
    for (int r = 0; r < 8; r++) creg[r] = cT[(gm.m0 + r) * HDIM + gm.hh];
    #pragma unroll
    for (int g = 0; g < 4; g++) {
      b_c[g] = bias[g * HDIM + gm.hh];
      for (int k = 0; k < FIN; k++) wih_c[g][k] = Wih[(g * HDIM + gm.hh) * FIN + k];
    }
  }
  float vwc[4];
  #pragma unroll
  for (int e = 0; e < 4; e++) vwc[e] = v_w[lane + 32 * e];

  const int gwave = blk * 32 + wave;            // 0..511
  const int gtid  = blk * 1024 + tid;           // 0..16383

  for (int t = 0; t < TSEQ; t++) {
    // ---- Phase A (wg0): LSTM cell + q = h @ W1^T -------------------------
    if (blk == 0) {
      __syncthreads();
      if (tid < BSZ * FIN) xtS[tid] = x[((tid / FIN) * TSEQ + t) * FIN + (tid % FIN)];
      ABu a[4];
      load_a_frags(hS, gm, a);
      __syncthreads();
      asm volatile("" ::: "memory");

      v8f acc[4];
      gate_gemm(WS, gm, n0, a, acc);

      Du dI, dF, dG, dO;
      dI.v = acc[0]; dF.v = acc[1]; dG.v = acc[2]; dO.v = acc[3];
      #pragma unroll
      for (int r = 0; r < 8; r++) {
        const int b = gm.m0 + r;
        const float* xr = &xtS[b * FIN];
        float s0 = b_c[0], s1 = b_c[1], s2 = b_c[2], s3 = b_c[3];
        #pragma unroll
        for (int k = 0; k < FIN; k++) {
          float xv = xr[k];
          s0 += xv * wih_c[0][k]; s1 += xv * wih_c[1][k];
          s2 += xv * wih_c[2][k]; s3 += xv * wih_c[3][k];
        }
        float zi = dI.f[r] + s0, zf = dF.f[r] + s1;
        float zg = dG.f[r] + s2, zo = dO.f[r] + s3;
        float cn = sigm(zf) * creg[r] + sigm(zi) * tanhf(zg);
        float hn = sigm(zo) * tanhf(cn);
        creg[r] = cn;
        hS[b * HDIM + gm.hh] = f2bf(hn);
        dec_cat[(b * TSEQ + t) * 2 * HDIM + gm.hh] = hn;     // concat: h part
      }
      __syncthreads();                       // new h visible for q GEMM
      {
        // q GEMM: 4 mtiles x 8 ntiles, one 16x16 tile per wave (same map)
        ABu a2[4];
        load_a_frags(hS, gm, a2);
        const unsigned short* wbase = W1B + (n0 * 16 + gm.ncol) * HDIM + gm.kB;
        v8f qa;
        #pragma unroll
        for (int r = 0; r < 8; r++) qa[r] = 0.0f;
        ABu bb0, bb1;
        load_b_frag(wbase, 0, bb0);
        #pragma unroll
        for (int kc = 0; kc < 4; kc++) {
          if (kc + 1 < 4) load_b_frag(wbase, kc + 1, (kc & 1) ? bb0 : bb1);
          qa = wmma_bf(a2[kc].v, (kc & 1) ? bb1.v : bb0.v, qa);
        }
        Du d; d.v = qa;
        #pragma unroll
        for (int r = 0; r < 8; r++)
          qbuf[(gm.m0 + r) * HDIM + n0 * 16 + gm.ncol] = d.f[r];
      }
    }
    gbar(cnt, gen, 16, phase);

    // ---- Phase B (all): score[b,t'] = sum_h v_w * tanh(q + enc_proj) -----
    #pragma unroll 1
    for (int s = 0; s < 16; s++) {
      int p = gwave * 16 + s, b = p >> 7, tp = p & 127;
      float acc = 0.0f;
      #pragma unroll
      for (int e = 0; e < 4; e++) {
        int hh = lane + 32 * e;
        float en = tanhf(qbuf[b * HDIM + hh] + enc_proj[(b * TSEQ + tp) * HDIM + hh]);
        acc += en * vwc[e];
      }
      #pragma unroll
      for (int off = 16; off > 0; off >>= 1) acc += __shfl_xor(acc, off, 32);
      if (lane == 0) score[b * TSEQ + tp] = acc;
    }
    gbar(cnt, gen, 16, phase);

    // ---- Phase C1: softmax over t' per batch row (64 waves) --------------
    if (gwave < BSZ) {
      int b = gwave;
      float v[4];
      #pragma unroll
      for (int e = 0; e < 4; e++) v[e] = score[b * TSEQ + lane + 32 * e];
      float mx = fmaxf(fmaxf(v[0], v[1]), fmaxf(v[2], v[3]));
      #pragma unroll
      for (int off = 16; off > 0; off >>= 1) mx = fmaxf(mx, __shfl_xor(mx, off, 32));
      float se = 0.0f;
      #pragma unroll
      for (int e = 0; e < 4; e++) { v[e] = __expf(v[e] - mx); se += v[e]; }
      #pragma unroll
      for (int off = 16; off > 0; off >>= 1) se += __shfl_xor(se, off, 32);
      float inv = 1.0f / se;
      #pragma unroll
      for (int e = 0; e < 4; e++) score[b * TSEQ + lane + 32 * e] = v[e] * inv;
    }
    gbar(cnt, gen, 16, phase);

    // ---- Phase C2: ctx[b,h] = sum_t' w[b,t'] * enc_out[b,t',h] -----------
    if (gtid < BSZ * HDIM) {
      int b = gtid >> 7, hh = gtid & 127;
      float acc = 0.0f;
      const float* eo = enc_out + (b * TSEQ) * HDIM + hh;
      const float* ww = score + b * TSEQ;
      for (int tp = 0; tp < TSEQ; tp++) acc += ww[tp] * eo[tp * HDIM];
      dec_cat[(b * TSEQ + t) * 2 * HDIM + HDIM + hh] = acc;  // concat: ctx part
    }
    gbar(cnt, gen, 16, phase);
  }
}

// ---------------------------------------------------------------------------
// out = dec_cat @ out_W^T + out_b     (8192 x 12, K=256)
// ---------------------------------------------------------------------------
__global__ void out_kernel(const float* __restrict__ dec_cat,
                           const float* __restrict__ out_W,
                           const float* __restrict__ out_b,
                           float* __restrict__ out) {
  int o = blockIdx.x * blockDim.x + threadIdx.x;
  if (o >= BSZ * TSEQ * FIN) return;
  int f = o % FIN, r = o / FIN;
  float acc = out_b[f];
  const float* dc = dec_cat + r * 2 * HDIM;
  const float* w  = out_W + f * 2 * HDIM;
  for (int j = 0; j < 2 * HDIM; j++) acc += dc[j] * w[j];
  out[o] = acc;
}

// ---------------------------------------------------------------------------
extern "C" void kernel_launch(void* const* d_in, const int* in_sizes, int n_in,
                              void* d_out, int out_size, void* d_ws, size_t ws_size,
                              hipStream_t stream) {
  (void)in_sizes; (void)n_in; (void)out_size; (void)ws_size;
  const float* x       = (const float*)d_in[0];
  const float* enc_Wih = (const float*)d_in[1];
  const float* enc_Whh = (const float*)d_in[2];
  const float* enc_b   = (const float*)d_in[3];
  const float* dec_Wih = (const float*)d_in[4];
  const float* dec_Whh = (const float*)d_in[5];
  const float* dec_b   = (const float*)d_in[6];
  const float* attn_W  = (const float*)d_in[7];
  const float* attn_b  = (const float*)d_in[8];
  const float* v_w     = (const float*)d_in[9];
  const float* out_W   = (const float*)d_in[10];
  const float* out_b   = (const float*)d_in[11];

  char* base = (char*)d_ws;
  size_t off = 0;
  auto alloc = [&](size_t bytes) -> void* {
    void* p = base + off;
    off = (off + bytes + 255) & ~(size_t)255;
    return p;
  };
  unsigned*       bar      = (unsigned*)alloc(256);
  float*          enc_out  = (float*)alloc((size_t)BSZ * TSEQ * HDIM * 4);       // 4 MB
  float*          enc_proj = (float*)alloc((size_t)BSZ * TSEQ * HDIM * 4);       // 4 MB
  float*          dec_cat  = (float*)alloc((size_t)BSZ * TSEQ * 2 * HDIM * 4);   // 8 MB
  float*          cT       = (float*)alloc((size_t)BSZ * HDIM * 4);
  float*          qbuf     = (float*)alloc((size_t)BSZ * HDIM * 4);
  float*          score    = (float*)alloc((size_t)BSZ * TSEQ * 4);
  unsigned short* WhhE     = (unsigned short*)alloc((size_t)GDIM * HDIM * 2);
  unsigned short* WhhD     = (unsigned short*)alloc((size_t)GDIM * HDIM * 2);
  unsigned short* W1B      = (unsigned short*)alloc((size_t)HDIM * HDIM * 2);
  unsigned short* W2B      = (unsigned short*)alloc((size_t)HDIM * HDIM * 2);

  prep_kernel<<<640, 256, 0, stream>>>(enc_Whh, dec_Whh, attn_W, WhhE, WhhD, W1B, W2B, bar);
  enc_kernel<<<1, 1024, 0, stream>>>(x, enc_Wih, enc_b, WhhE, enc_out, cT);
  proj_kernel<<<512, 256, 0, stream>>>(enc_out, W2B, attn_b, enc_proj);
  dec_kernel<<<16, 1024, 0, stream>>>(x, dec_Wih, dec_b, WhhD, W1B, enc_out, enc_proj,
                                      cT, v_w, qbuf, score, dec_cat, bar);
  out_kernel<<<384, 256, 0, stream>>>(dec_cat, out_W, out_b, (float*)d_out);
}
